// EGCN_60249801228790
// MI455X (gfx1250) — matrix operand order repaired and checked by reading, not compiled, not run on hardware
//
#include <hip/hip_runtime.h>
#include <hip/hip_bf16.h>

// ---------------------------------------------------------------------------
// EGNN forward for MI455X (gfx1250, wave32, WMMA).
// Edge MLP chain mapped to v_wmma_f32_16x16x32_f16, 16 edges per wave.
// Edge-BN folded into the first edge-MLP weight (row scale + effective bias).
// Node tables (6.4MB) are L2-resident; scatter via global_atomic_add_f32.
// All fragment loads vectorized to b128 (global + LDS).
// ---------------------------------------------------------------------------

typedef __attribute__((ext_vector_type(16))) _Float16 v16h;
typedef __attribute__((ext_vector_type(8)))  _Float16 v8h;
typedef __attribute__((ext_vector_type(8)))  float    v8f;
typedef __attribute__((ext_vector_type(4)))  float    v4f;

#define BN_EPS 1e-5f

__device__ __forceinline__ float lrelu(float x) { return x > 0.0f ? x : 0.01f * x; }

__device__ __forceinline__ v8f wmma16(v16h a, v16h b, v8f c) {
  // D = A(16x32 f16) * B(32x16 f16) + C(16x16 f32)
  return __builtin_amdgcn_wmma_f32_16x16x32_f16(false, a, false, b, (short)0, c, false, false);
}

// B fragment: 32x16 f16 tile of a KxNmat row-major matrix.
// Layout (ISA 7.12.2): lane n=L&15 holds column ntile*16+n; half h=L>>4
// holds K = 8h + i for i<8, K = 16 + 8h + (i-8) for i>=8.
__device__ __forceinline__ v16h load_bfrag(const _Float16* __restrict__ W, int ldb,
                                           int kbase, int ntile, int lane) {
  const int h = lane >> 4, n = lane & 15;
  v16h b;
#pragma unroll
  for (int i = 0; i < 16; ++i)
    b[i] = W[(kbase + 8 * h + i + (i >= 8 ? 8 : 0)) * ldb + ntile * 16 + n];
  return b;
}

// A fragment from a row-major f32 row pointer (32 cols, 16B-aligned rows).
// A 16x32 f16 layout: i<8 -> K=8h+i ; i>=8 -> K=16+8h+(i-8).
__device__ __forceinline__ v16h load_afrag_f32row(const float* __restrict__ row, int h) {
  const v4f* r4 = (const v4f*)row;
  v4f p0 = r4[2 * h], p1 = r4[2 * h + 1], p2 = r4[4 + 2 * h], p3 = r4[5 + 2 * h];
  v16h a;
#pragma unroll
  for (int i = 0; i < 4; ++i) {
    a[i]      = (_Float16)p0[i];
    a[4 + i]  = (_Float16)p1[i];
    a[8 + i]  = (_Float16)p2[i];
    a[12 + i] = (_Float16)p3[i];
  }
  return a;
}

// A fragment from a 16x32 f16 tile in LDS (D->A relayout read, b128 loads).
__device__ __forceinline__ v16h load_afrag_lds(const _Float16* sm, int mrow, int h) {
  v8h lo = *(const v8h*)&sm[mrow * 32 + 8 * h];
  v8h hi = *(const v8h*)&sm[mrow * 32 + 16 + 8 * h];
  v16h a;
#pragma unroll
  for (int i = 0; i < 8; ++i) { a[i] = lo[i]; a[8 + i] = hi[i]; }
  return a;
}

// ---------------------------------------------------------------------------
// Small elementwise / reduction kernels
// ---------------------------------------------------------------------------

// z[n,c] = sum_k [h|x][n,k] * Wr[k,c] + br[c]   (N x 8 @ 8 x 64)
__global__ void k_zr(const float* __restrict__ h, const float* __restrict__ x,
                     const float* __restrict__ Wr, const float* __restrict__ br,
                     float* __restrict__ z, int N) {
  int id = blockIdx.x * blockDim.x + threadIdx.x;
  if (id >= N * 64) return;
  int n = id >> 6, c = id & 63;
  float acc = br[c];
#pragma unroll
  for (int k = 0; k < 6; ++k) acc += h[n * 6 + k] * Wr[k * 64 + c];
#pragma unroll
  for (int k = 0; k < 2; ++k) acc += x[n * 2 + k] * Wr[(6 + k) * 64 + c];
  z[id] = acc;
}

// Per-column sum / sumsq, optional integer weights (for degree-weighted stats).
__global__ void k_colstats(const float* __restrict__ X, const int* __restrict__ wgt,
                           int rows, int cols,
                           float* __restrict__ out_sum, float* __restrict__ out_sumsq) {
  const int c = blockIdx.x;
  float s = 0.f, ss = 0.f;
  for (int r = blockIdx.y * blockDim.x + threadIdx.x; r < rows;
       r += gridDim.y * blockDim.x) {
    float v = X[(long long)r * cols + c];
    float w = wgt ? (float)wgt[r] : 1.0f;
    s += w * v;
    ss += w * v * v;
  }
  __shared__ float rs[256], rss[256];
  rs[threadIdx.x] = s; rss[threadIdx.x] = ss;
  __syncthreads();
  for (int o = 128; o > 0; o >>= 1) {
    if (threadIdx.x < o) { rs[threadIdx.x] += rs[threadIdx.x + o]; rss[threadIdx.x] += rss[threadIdx.x + o]; }
    __syncthreads();
  }
  if (threadIdx.x == 0) { atomicAdd(&out_sum[c], rs[0]); atomicAdd(&out_sumsq[c], rss[0]); }
}

// z -> lrelu(bn(z)) -> split into hh (cols 0..31) / xx (cols 32..63)
__global__ void k_apply_init(const float* __restrict__ z, const float* __restrict__ g,
                             const float* __restrict__ b, const float* __restrict__ zbn,
                             float* __restrict__ hh, float* __restrict__ xx, int N) {
  int id = blockIdx.x * blockDim.x + threadIdx.x;
  if (id >= N * 64) return;
  int n = id >> 6, c = id & 63;
  float inv = 1.0f / (float)N;
  float mu = zbn[c] * inv;
  float var = zbn[64 + c] * inv - mu * mu;
  float v = lrelu((z[id] - mu) * rsqrtf(var + BN_EPS) * g[c] + b[c]);
  if (c < 32) hh[n * 32 + c] = v; else xx[n * 32 + (c - 32)] = v;
}

__global__ void k_deg(const int* __restrict__ ei, long long E,
                      int* __restrict__ deg_src, int* __restrict__ deg_dst) {
  long long e = (long long)blockIdx.x * blockDim.x + threadIdx.x;
  if (e >= E) return;
  atomicAdd(&deg_src[ei[e]], 1);
  atomicAdd(&deg_dst[ei[E + e]], 1);
}

// y = lrelu(bn(hh))
__global__ void k_y(const float* __restrict__ hh, const float* __restrict__ g,
                    const float* __restrict__ b, const float* __restrict__ nbn,
                    float* __restrict__ y, int N) {
  int id = blockIdx.x * blockDim.x + threadIdx.x;
  if (id >= N * 32) return;
  int c = id & 31;
  float inv = 1.0f / (float)N;
  float mu = nbn[c] * inv;
  float var = nbn[32 + c] * inv - mu * mu;
  y[id] = lrelu((hh[id] - mu) * rsqrtf(var + BN_EPS) * g[c] + b[c]);
}

// sum / sumsq of ||xx[dst]-xx[src]|| over edges (edge-BN column 64)
__global__ void k_diststats(const int* __restrict__ ei, long long E,
                            const float* __restrict__ xx,
                            float* __restrict__ out_sum, float* __restrict__ out_sumsq) {
  float s = 0.f, ss = 0.f;
  for (long long e = (long long)blockIdx.x * blockDim.x + threadIdx.x; e < E;
       e += (long long)gridDim.x * blockDim.x) {
    const v4f* xd = (const v4f*)(xx + (long long)ei[E + e] * 32);
    const v4f* xs = (const v4f*)(xx + (long long)ei[e] * 32);
    float d2 = 0.f;
#pragma unroll
    for (int q = 0; q < 8; ++q) {
      v4f dv = xd[q] - xs[q];
      d2 += dv[0] * dv[0] + dv[1] * dv[1] + dv[2] * dv[2] + dv[3] * dv[3];
    }
    s += __builtin_amdgcn_sqrtf(d2);
    ss += d2;
  }
  __shared__ float rs[256], rss[256];
  rs[threadIdx.x] = s; rss[threadIdx.x] = ss;
  __syncthreads();
  for (int o = 128; o > 0; o >>= 1) {
    if (threadIdx.x < o) { rs[threadIdx.x] += rs[threadIdx.x + o]; rss[threadIdx.x] += rss[threadIdx.x + o]; }
    __syncthreads();
  }
  if (threadIdx.x == 0) { atomicAdd(out_sum, rs[0]); atomicAdd(out_sumsq, rss[0]); }
}

// ew1s[k,j] = (rs_k * eg_k) * ew1[k,j] for k<65 (f16, K padded to 96 with zeros)
__global__ void k_prep_ew1(const float* __restrict__ ew1, const float* __restrict__ eg,
                           const float* __restrict__ ebn, float Ef,
                           _Float16* __restrict__ ew1s) {
  int id = blockIdx.x * blockDim.x + threadIdx.x;
  if (id >= 96 * 32) return;
  int k = id >> 5, j = id & 31;
  _Float16 v = (_Float16)0.0f;
  if (k < 65) {
    float mu = ebn[k] / Ef;
    float var = ebn[65 + k] / Ef - mu * mu;
    float S = rsqrtf(var + BN_EPS) * eg[k];
    v = (_Float16)(S * ew1[k * 32 + j]);
  }
  ew1s[id] = v;
}

// eb1eff[j] = eb1[j] + sum_k (ebeta_k - mu_k * rs_k * eg_k) * ew1[k,j]
__global__ void k_eb1eff(const float* __restrict__ ew1, const float* __restrict__ eb1,
                         const float* __restrict__ eg, const float* __restrict__ ebeta,
                         const float* __restrict__ ebn, float Ef,
                         float* __restrict__ out) {
  int j = threadIdx.x;
  if (j >= 32) return;
  float acc = eb1[j];
  for (int k = 0; k < 65; ++k) {
    float mu = ebn[k] / Ef;
    float var = ebn[65 + k] / Ef - mu * mu;
    float S = rsqrtf(var + BN_EPS) * eg[k];
    acc += (ebeta[k] - mu * S) * ew1[k * 32 + j];
  }
  out[j] = acc;
}

__global__ void k_packf16(const float* __restrict__ src, _Float16* __restrict__ dst, int count) {
  int id = blockIdx.x * blockDim.x + threadIdx.x;
  if (id < count) dst[id] = (_Float16)src[id];
}

// out = hh @ Wo + bo  (32 -> 2)
__global__ void k_out(const float* __restrict__ hh, const float* __restrict__ Wo,
                      const float* __restrict__ bo, float* __restrict__ out, int N) {
  int n = blockIdx.x * blockDim.x + threadIdx.x;
  if (n >= N) return;
  float a0 = bo[0], a1 = bo[1];
#pragma unroll
  for (int c = 0; c < 32; ++c) {
    float hv = hh[n * 32 + c];
    a0 += hv * Wo[c * 2 + 0];
    a1 += hv * Wo[c * 2 + 1];
  }
  out[n * 2 + 0] = a0;
  out[n * 2 + 1] = a1;
}

// ---------------------------------------------------------------------------
// Edge kernel: 16 edges per wave, WMMA MLP chain, fused scatter-add.
// ---------------------------------------------------------------------------
__global__ void __launch_bounds__(256) k_edge(
    const int* __restrict__ ei, const long long E,
    const float* __restrict__ y, const float* __restrict__ xx,
    const _Float16* __restrict__ ew1s, const float* __restrict__ eb1eff,
    const _Float16* __restrict__ ew2f, const float* __restrict__ eb2,
    const _Float16* __restrict__ cw1f, const float* __restrict__ cb1,
    const _Float16* __restrict__ cw2f,
    float* __restrict__ m_agg, float* __restrict__ x_n, const int ntiles) {
  __shared__ int s_src[8][16];
  __shared__ int s_dst[8][16];
  __shared__ __align__(16) float    s_dx[8][512];  // 16 rows x 32 cols per wave
  __shared__ __align__(16) _Float16 s_m[8][512];   // D-layout -> A-layout bounce

  const int lane = threadIdx.x & 31;
  const int wid  = threadIdx.x >> 5;
  const int mrow = lane & 15;            // A-matrix row
  const int h    = lane >> 4;            // lane half
  const int n    = mrow;                 // output column within 16-wide tile

  // Resident weight fragments (all matrices tiny -> keep in VGPRs).
  v16h Bw1[3][2];
#pragma unroll
  for (int c = 0; c < 3; ++c)
#pragma unroll
    for (int t = 0; t < 2; ++t)
      Bw1[c][t] = load_bfrag(ew1s, 32, 32 * c, t, lane);
  v16h Bw2[2], Bc1[2], Bc2[2];
#pragma unroll
  for (int t = 0; t < 2; ++t) {
    Bw2[t] = load_bfrag(ew2f, 32, 0, t, lane);
    Bc1[t] = load_bfrag(cw1f, 32, 0, t, lane);
    Bc2[t] = load_bfrag(cw2f, 32, 0, t, lane);
  }
  const float be1_0 = eb1eff[n], be1_1 = eb1eff[16 + n];
  const float be2_0 = eb2[n],    be2_1 = eb2[16 + n];
  const float bc1_0 = cb1[n],    bc1_1 = cb1[16 + n];

  for (int tile = blockIdx.x * 8 + wid; tile < ntiles; tile += gridDim.x * 8) {
    const long long e  = (long long)tile * 16 + mrow;
    const long long ec = e < E ? e : E - 1;
    const int si = ei[ec];
    const int di = ei[E + ec];
    __builtin_prefetch(&ei[ec + 16 * 8 * (long long)gridDim.x], 0, 3);
    s_src[wid][mrow] = si;
    s_dst[wid][mrow] = di;

    // Stage dx = xx[dst] - xx[src]; one column per lane per row.
#pragma unroll
    for (int r = 0; r < 16; ++r) {
      const int dn = s_dst[wid][r];
      const int sn = s_src[wid][r];
      s_dx[wid][r * 32 + lane] = xx[dn * 32 + lane] - xx[sn * 32 + lane];
    }

    // dist of own row (b128 LDS reads + raw v_sqrt_f32)
    float d2 = 0.f;
    {
      const v4f* dxr = (const v4f*)&s_dx[wid][mrow * 32];
#pragma unroll
      for (int q = 0; q < 8; ++q) {
        v4f v = dxr[q];
        d2 += v[0] * v[0] + v[1] * v[1] + v[2] * v[2] + v[3] * v[3];
      }
    }
    const float dist = __builtin_amdgcn_sqrtf(d2);

    // A fragments: raw features (edge BN folded into ew1s/eb1eff).
    v16h a0 = load_afrag_f32row(y + (long long)di * 32, h);
    v16h a1 = load_afrag_f32row(y + (long long)si * 32, h);
    v16h a2 = {};
    if (h == 0) a2[0] = (_Float16)dist;  // K=64 (dist), rest of chunk is zero pad

    // GEMM 1: (16x96) @ (96x32) -> 16x32, bias + lrelu
    v8f t0 = {}, t1 = {};
    t0 = wmma16(a0, Bw1[0][0], t0); t1 = wmma16(a0, Bw1[0][1], t1);
    t0 = wmma16(a1, Bw1[1][0], t0); t1 = wmma16(a1, Bw1[1][1], t1);
    t0 = wmma16(a2, Bw1[2][0], t0); t1 = wmma16(a2, Bw1[2][1], t1);
#pragma unroll
    for (int v = 0; v < 8; ++v) { t0[v] = lrelu(t0[v] + be1_0); t1[v] = lrelu(t1[v] + be1_1); }

    // D-layout -> A-layout via LDS
#pragma unroll
    for (int v = 0; v < 8; ++v) {
      s_m[wid][(v + 8 * h) * 32 + n]      = (_Float16)t0[v];
      s_m[wid][(v + 8 * h) * 32 + 16 + n] = (_Float16)t1[v];
    }
    v16h am = load_afrag_lds(s_m[wid], mrow, h);

    // GEMM 2: -> m (message), bias + lrelu
    v8f m0 = {}, m1 = {};
    m0 = wmma16(am, Bw2[0], m0); m1 = wmma16(am, Bw2[1], m1);
#pragma unroll
    for (int v = 0; v < 8; ++v) { m0[v] = lrelu(m0[v] + be2_0); m1[v] = lrelu(m1[v] + be2_1); }

    // GEMM 3: gate hidden = lrelu(m @ cw1 + cb1)
#pragma unroll
    for (int v = 0; v < 8; ++v) {
      s_m[wid][(v + 8 * h) * 32 + n]      = (_Float16)m0[v];
      s_m[wid][(v + 8 * h) * 32 + 16 + n] = (_Float16)m1[v];
    }
    v16h ag = load_afrag_lds(s_m[wid], mrow, h);
    v8f g0 = {}, g1 = {};
    g0 = wmma16(ag, Bc1[0], g0); g1 = wmma16(ag, Bc1[1], g1);
#pragma unroll
    for (int v = 0; v < 8; ++v) { g0[v] = lrelu(g0[v] + bc1_0); g1[v] = lrelu(g1[v] + bc1_1); }

    // GEMM 4: gate = hidden @ cw2 (no bias, no activation)
#pragma unroll
    for (int v = 0; v < 8; ++v) {
      s_m[wid][(v + 8 * h) * 32 + n]      = (_Float16)g0[v];
      s_m[wid][(v + 8 * h) * 32 + 16 + n] = (_Float16)g1[v];
    }
    v16h aq = load_afrag_lds(s_m[wid], mrow, h);
    v8f q0 = {}, q1 = {};
    q0 = wmma16(aq, Bc2[0], q0); q1 = wmma16(aq, Bc2[1], q1);

    // Fused scatter: segment_sum(m) and segment_sum(dx * gate) via f32 atomics.
    // D layout: VGPR v of lane (h,n) holds (row v+8h, col 16t+n).
#pragma unroll
    for (int v = 0; v < 8; ++v) {
      const int r = v + 8 * h;
      if ((long long)tile * 16 + r < E) {
        const int dn = s_dst[wid][r];
        atomicAdd(&m_agg[dn * 32 + n],      m0[v]);
        atomicAdd(&m_agg[dn * 32 + 16 + n], m1[v]);
        atomicAdd(&x_n[dn * 32 + n],      s_dx[wid][r * 32 + n]      * q0[v]);
        atomicAdd(&x_n[dn * 32 + 16 + n], s_dx[wid][r * 32 + 16 + n] * q1[v]);
      }
    }
  }
}

// ---------------------------------------------------------------------------
// Node update: h_n = lrelu([y|m_agg] @ nw1 + nb1) @ nw2 + nb2 ; hh += ; xx = x_n
// ---------------------------------------------------------------------------
__global__ void __launch_bounds__(256) k_node(
    const float* __restrict__ y, const float* __restrict__ m_agg,
    const float* __restrict__ x_n,
    const _Float16* __restrict__ nw1f, const float* __restrict__ nb1,
    const _Float16* __restrict__ nw2f, const float* __restrict__ nb2,
    float* __restrict__ hh, float* __restrict__ xx, const int N, const int ntiles) {
  __shared__ __align__(16) _Float16 s_m[8][512];

  const int lane = threadIdx.x & 31;
  const int wid  = threadIdx.x >> 5;
  const int mrow = lane & 15;
  const int h    = lane >> 4;
  const int n    = mrow;

  v16h B1[2][2], B2[2];
#pragma unroll
  for (int c = 0; c < 2; ++c)
#pragma unroll
    for (int t = 0; t < 2; ++t)
      B1[c][t] = load_bfrag(nw1f, 32, 32 * c, t, lane);
#pragma unroll
  for (int t = 0; t < 2; ++t) B2[t] = load_bfrag(nw2f, 32, 0, t, lane);
  const float b1_0 = nb1[n], b1_1 = nb1[16 + n];
  const float b2_0 = nb2[n], b2_1 = nb2[16 + n];

  for (int tile = blockIdx.x * 8 + wid; tile < ntiles; tile += gridDim.x * 8) {
    int row = tile * 16 + mrow;
    int rc = row < N ? row : N - 1;
    v16h a0 = load_afrag_f32row(y + (long long)rc * 32, h);
    v16h a1 = load_afrag_f32row(m_agg + (long long)rc * 32, h);

    v8f t0 = {}, t1 = {};
    t0 = wmma16(a0, B1[0][0], t0); t1 = wmma16(a0, B1[0][1], t1);
    t0 = wmma16(a1, B1[1][0], t0); t1 = wmma16(a1, B1[1][1], t1);
#pragma unroll
    for (int v = 0; v < 8; ++v) { t0[v] = lrelu(t0[v] + b1_0); t1[v] = lrelu(t1[v] + b1_1); }

#pragma unroll
    for (int v = 0; v < 8; ++v) {
      s_m[wid][(v + 8 * h) * 32 + n]      = (_Float16)t0[v];
      s_m[wid][(v + 8 * h) * 32 + 16 + n] = (_Float16)t1[v];
    }
    v16h am = load_afrag_lds(s_m[wid], mrow, h);

    v8f h0 = {}, h1 = {};
    h0 = wmma16(am, B2[0], h0); h1 = wmma16(am, B2[1], h1);

#pragma unroll
    for (int v = 0; v < 8; ++v) {
      const int node = tile * 16 + v + 8 * h;
      if (node < N) {
        hh[node * 32 + n]      += h0[v] + b2_0;
        hh[node * 32 + 16 + n] += h1[v] + b2_1;
        xx[node * 32 + n]      = x_n[node * 32 + n];
        xx[node * 32 + 16 + n] = x_n[node * 32 + 16 + n];
      }
    }
  }
}

// ---------------------------------------------------------------------------
// Host-side orchestration
// ---------------------------------------------------------------------------
extern "C" void kernel_launch(void* const* d_in, const int* in_sizes, int n_in,
                              void* d_out, int out_size, void* d_ws, size_t ws_size,
                              hipStream_t stream) {
  const float* h_in  = (const float*)d_in[0];
  const float* x_in  = (const float*)d_in[1];
  const int*   ei    = (const int*)d_in[2];
  const float* Wr    = (const float*)d_in[3];
  const float* br    = (const float*)d_in[4];
  const float* gr    = (const float*)d_in[5];
  const float* betar = (const float*)d_in[6];
  const float* bn_g  = (const float*)d_in[7];
  const float* bn_b  = (const float*)d_in[8];
  const float* eg    = (const float*)d_in[9];
  const float* ebeta = (const float*)d_in[10];
  const float* ew1   = (const float*)d_in[11];
  const float* eb1   = (const float*)d_in[12];
  const float* ew2   = (const float*)d_in[13];
  const float* eb2   = (const float*)d_in[14];
  const float* cw1   = (const float*)d_in[15];
  const float* cb1   = (const float*)d_in[16];
  const float* cw2   = (const float*)d_in[17];
  const float* nw1   = (const float*)d_in[18];
  const float* nb1   = (const float*)d_in[19];
  const float* nw2   = (const float*)d_in[20];
  const float* nb2   = (const float*)d_in[21];
  const float* Wo    = (const float*)d_in[22];
  const float* bo    = (const float*)d_in[23];
  float* out = (float*)d_out;

  const int N = in_sizes[0] / 6;
  const long long E = (long long)in_sizes[2] / 2;

  // workspace carve-out
  char* p = (char*)d_ws;
  auto alloc = [&](size_t bytes) -> void* {
    void* r = (void*)p;
    p += (bytes + 255) & ~(size_t)255;
    return r;
  };
  float* z       = (float*)alloc((size_t)N * 64 * 4);
  float* hh      = (float*)alloc((size_t)N * 32 * 4);
  float* xx      = (float*)alloc((size_t)N * 32 * 4);
  float* y       = (float*)alloc((size_t)N * 32 * 4);
  float* m_agg   = (float*)alloc((size_t)N * 32 * 4);
  float* x_n     = (float*)alloc((size_t)N * 32 * 4);
  int*   deg_src = (int*)alloc((size_t)N * 4);
  int*   deg_dst = (int*)alloc((size_t)N * 4);
  float* zbn     = (float*)alloc(128 * 4);
  float* nbn     = (float*)alloc(64 * 4);
  float* ebn     = (float*)alloc(130 * 4);
  _Float16* ew1s   = (_Float16*)alloc(96 * 32 * 2);
  float*    eb1eff = (float*)alloc(32 * 4);
  _Float16* ew2f   = (_Float16*)alloc(1024 * 2);
  _Float16* cw1f   = (_Float16*)alloc(1024 * 2);
  _Float16* cw2f   = (_Float16*)alloc(1024 * 2);
  _Float16* nw1f   = (_Float16*)alloc(2048 * 2);
  _Float16* nw2f   = (_Float16*)alloc(1024 * 2);

  // ---- initial projection + BN ----
  hipMemsetAsync(zbn, 0, 128 * 4, stream);
  hipMemsetAsync(deg_src, 0, (size_t)N * 4, stream);
  hipMemsetAsync(deg_dst, 0, (size_t)N * 4, stream);

  k_zr<<<(N * 64 + 255) / 256, 256, 0, stream>>>(h_in, x_in, Wr, br, z, N);
  k_colstats<<<dim3(64, 64), 256, 0, stream>>>(z, nullptr, N, 64, zbn, zbn + 64);
  k_apply_init<<<(N * 64 + 255) / 256, 256, 0, stream>>>(z, gr, betar, zbn, hh, xx, N);
  k_deg<<<(int)((E + 255) / 256), 256, 0, stream>>>(ei, E, deg_src, deg_dst);

  const int ntiles_e = (int)((E + 15) / 16);
  const int ntiles_n = (N + 15) / 16;
  const int blocks_e = 1536;
  const int blocks_n = (ntiles_n + 7) / 8;

  for (int l = 0; l < 2; ++l) {
    const float* eg_l    = eg + l * 65;
    const float* ebeta_l = ebeta + l * 65;
    const float* ew1_l   = ew1 + l * 65 * 32;
    const float* eb1_l   = eb1 + l * 32;
    const float* ew2_l   = ew2 + l * 1024;
    const float* eb2_l   = eb2 + l * 32;
    const float* cw1_l   = cw1 + l * 1024;
    const float* cb1_l   = cb1 + l * 32;
    const float* cw2_l   = cw2 + l * 1024;
    const float* nw1_l   = nw1 + l * 2048;
    const float* nb1_l   = nb1 + l * 32;
    const float* nw2_l   = nw2 + l * 1024;
    const float* nb2_l   = nb2 + l * 32;

    hipMemsetAsync(nbn, 0, 64 * 4, stream);
    hipMemsetAsync(ebn, 0, 130 * 4, stream);
    hipMemsetAsync(m_agg, 0, (size_t)N * 32 * 4, stream);
    hipMemsetAsync(x_n, 0, (size_t)N * 32 * 4, stream);

    // node BN -> y
    k_colstats<<<dim3(32, 64), 256, 0, stream>>>(hh, nullptr, N, 32, nbn, nbn + 32);
    k_y<<<(N * 32 + 255) / 256, 256, 0, stream>>>(hh, bn_g + l * 32, bn_b + l * 32, nbn, y, N);

    // edge-BN stats: degree-weighted y sums (O(N)) + dist pass (O(E))
    k_colstats<<<dim3(32, 64), 256, 0, stream>>>(y, deg_dst, N, 32, ebn + 0,  ebn + 65 + 0);
    k_colstats<<<dim3(32, 64), 256, 0, stream>>>(y, deg_src, N, 32, ebn + 32, ebn + 65 + 32);
    k_diststats<<<512, 256, 0, stream>>>(ei, E, xx, ebn + 64, ebn + 65 + 64);

    // fold edge BN into ew1 + effective bias; pack f16 weights
    k_prep_ew1<<<(96 * 32 + 255) / 256, 256, 0, stream>>>(ew1_l, eg_l, ebn, (float)E, ew1s);
    k_eb1eff<<<1, 32, 0, stream>>>(ew1_l, eb1_l, eg_l, ebeta_l, ebn, (float)E, eb1eff);
    k_packf16<<<4, 256, 0, stream>>>(ew2_l, ew2f, 1024);
    k_packf16<<<4, 256, 0, stream>>>(cw1_l, cw1f, 1024);
    k_packf16<<<4, 256, 0, stream>>>(cw2_l, cw2f, 1024);
    k_packf16<<<8, 256, 0, stream>>>(nw1_l, nw1f, 2048);
    k_packf16<<<4, 256, 0, stream>>>(nw2_l, nw2f, 1024);

    // fused WMMA edge MLP + scatter, then WMMA node update
    k_edge<<<blocks_e, 256, 0, stream>>>(ei, E, y, xx, ew1s, eb1eff, ew2f, eb2_l,
                                         cw1f, cb1_l, cw2f, m_agg, x_n, ntiles_e);
    k_node<<<blocks_n, 256, 0, stream>>>(y, m_agg, x_n, nw1f, nb1_l, nw2f, nb2_l,
                                         hh, xx, N, ntiles_n);
  }

  k_out<<<(N + 255) / 256, 256, 0, stream>>>(hh, Wo, bo, out, N);
}
